// CFDSurrogateModel_83949430768017
// MI455X (gfx1250) — matrix operand order repaired and checked by reading, not compiled, not run on hardware
//
#include <hip/hip_runtime.h>
#include <hip/hip_bf16.h>
#include <math.h>

// ---------------------------------------------------------------------------
// CFD GNN surrogate, MI455X (gfx1250).  f16 WMMA GEMMs (v_wmma_f32_16x16x32_f16),
// in-register LayerNorm via half-wave shuffles, LDS-staged activations,
// f32 atomics for the segment-sum aggregation.
// ---------------------------------------------------------------------------

typedef __attribute__((ext_vector_type(16))) _Float16 v16h;
typedef __attribute__((ext_vector_type(8)))  _Float16 v8h;
typedef __attribute__((ext_vector_type(8)))  float    v8f;

#define DEV __device__ __forceinline__

static constexpr int Nn = 50000;
static constexpr int Ee = 600000;
static constexpr int Hh = 128;
static constexpr int Ll = 10;

DEV v8f v8f_zero() { v8f z = {0.f,0.f,0.f,0.f,0.f,0.f,0.f,0.f}; return z; }

DEV v8f wmma16(v16h a, v16h b, v8f c) {
  return __builtin_amdgcn_wmma_f32_16x16x32_f16(
      /*neg_a=*/false, a, /*neg_b=*/false, b,
      /*c_mod=*/(short)0, c, /*reuse_a=*/false, /*reuse_b=*/false);
}

DEV float gelu_f(float x) { return 0.5f * x * (1.0f + erff(x * 0.70710678118654752f)); }

// sum across the 16 lanes of a half-wave (lanes 0-15 or 16-31)
DEV float hsum16(float v) {
  v += __shfl_xor(v, 1, 32);
  v += __shfl_xor(v, 2, 32);
  v += __shfl_xor(v, 4, 32);
  v += __shfl_xor(v, 8, 32);
  return v;
}

// A fragment, 16x32 f16 (rows mb..mb+15, K k0..k0+31), row-major source, stride ldk halves.
// ISA layout: lanes 0-15 hold K {0..7,16..23}; lanes 16-31 hold K {8..15,24..31}.
DEV v16h load_frag_a(const _Float16* base, int ldk, int mb, int k0) {
  const int lane = threadIdx.x & 31;
  const _Float16* p = base + (size_t)(mb + (lane & 15)) * ldk + k0 + ((lane >> 4) << 3);
  v8h lo = *(const v8h*)(p);
  v8h hi = *(const v8h*)(p + 16);
  v16h r;
#pragma unroll
  for (int i = 0; i < 8; ++i) { r[i] = lo[i]; r[i + 8] = hi[i]; }
  return r;
}

// B fragment, 32x16 f16 (K k0..k0+31, cols n0..n0+15) from W^T stored [nout][K].
// ISA layout: lanes 0-15 hold K {0..7,16..23}; lanes 16-31 hold K {8..15,24..31}.
DEV v16h load_frag_b(const _Float16* base, int ldk, int n0, int k0) {
  const int lane = threadIdx.x & 31;
  const _Float16* p = base + (size_t)(n0 + (lane & 15)) * ldk + k0 + ((lane >> 4) << 3);
  v8h lo = *(const v8h*)(p);
  v8h hi = *(const v8h*)(p + 16);
  v16h r;
#pragma unroll
  for (int i = 0; i < 8; ++i) { r[i] = lo[i]; r[i + 8] = hi[i]; }
  return r;
}

// Fused: z = LN2( GELU(LN1(A @ W1 + b1)) @ W2 + b2 )
// A is 64 x K1 f16 in `lds`; each of the 4 waves owns rows wave*16..wave*16+15 and ALL
// output columns, so LayerNorm reduces in-register across 16 lanes.  The LDS buffer is
// reused (with syncs) for the GEMM2 input (64 x N1 f16).
template <int K1, int N1, int N2>
DEV void fused_mlp(_Float16* lds,
                   const _Float16* __restrict__ w1t, const float* __restrict__ b1,
                   const float* __restrict__ g1, const float* __restrict__ bb1,
                   const _Float16* __restrict__ w2t, const float* __restrict__ b2,
                   const float* __restrict__ g2, const float* __restrict__ bb2,
                   v8f (&acc2)[N2 / 16]) {
  const int lane = threadIdx.x & 31;
  const int wave = threadIdx.x >> 5;
  const int mb = wave * 16;
  constexpr int NT1 = N1 / 16, NT2 = N2 / 16, KS1 = K1 / 32, KS2 = N1 / 32;

  // ---- GEMM1: [64,K1] x [K1,N1]
  v8f acc1[NT1];
#pragma unroll
  for (int nt = 0; nt < NT1; ++nt) acc1[nt] = v8f_zero();
  for (int ks = 0; ks < KS1; ++ks) {
    v16h a = load_frag_a(lds, K1, mb, ks * 32);
#pragma unroll
    for (int nt = 0; nt < NT1; ++nt) {
      v16h b = load_frag_b(w1t, K1, nt * 16, ks * 32);
      acc1[nt] = wmma16(a, b, acc1[nt]);
    }
  }
  // bias
#pragma unroll
  for (int nt = 0; nt < NT1; ++nt) {
    float bias = b1[nt * 16 + (lane & 15)];
#pragma unroll
    for (int r = 0; r < 8; ++r) acc1[nt][r] += bias;
  }
  // LayerNorm stats per row (row r (+8 for upper half) spread across 16 lanes x NT1 tiles)
  float mean[8], rstd[8];
#pragma unroll
  for (int r = 0; r < 8; ++r) {
    float p = 0.f, q = 0.f;
#pragma unroll
    for (int nt = 0; nt < NT1; ++nt) { float v = acc1[nt][r]; p += v; q += v * v; }
    p = hsum16(p); q = hsum16(q);
    float mu = p * (1.0f / N1);
    mean[r] = mu;
    rstd[r] = rsqrtf(q * (1.0f / N1) - mu * mu + 1e-5f);
  }
  __syncthreads();  // everyone done reading A1 region before aliasing it as A2
  // normalize + GELU + store GEMM2 input (f16) to LDS
#pragma unroll
  for (int nt = 0; nt < NT1; ++nt) {
    int n = nt * 16 + (lane & 15);
    float g = g1[n], bb = bb1[n];
#pragma unroll
    for (int r = 0; r < 8; ++r) {
      float v = (acc1[nt][r] - mean[r]) * rstd[r] * g + bb;
      v = gelu_f(v);
      int m = mb + r + ((lane >> 4) << 3);
      lds[(size_t)m * N1 + n] = (_Float16)v;
    }
  }
  __syncthreads();

  // ---- GEMM2: [64,N1] x [N1,N2]
#pragma unroll
  for (int nt = 0; nt < NT2; ++nt) acc2[nt] = v8f_zero();
  for (int ks = 0; ks < KS2; ++ks) {
    v16h a = load_frag_a(lds, N1, mb, ks * 32);
#pragma unroll
    for (int nt = 0; nt < NT2; ++nt) {
      v16h b = load_frag_b(w2t, N1, nt * 16, ks * 32);
      acc2[nt] = wmma16(a, b, acc2[nt]);
    }
  }
#pragma unroll
  for (int nt = 0; nt < NT2; ++nt) {
    float bias = b2[nt * 16 + (lane & 15)];
#pragma unroll
    for (int r = 0; r < 8; ++r) acc2[nt][r] += bias;
  }
#pragma unroll
  for (int r = 0; r < 8; ++r) {
    float p = 0.f, q = 0.f;
#pragma unroll
    for (int nt = 0; nt < NT2; ++nt) { float v = acc2[nt][r]; p += v; q += v * v; }
    p = hsum16(p); q = hsum16(q);
    float mu = p * (1.0f / N2);
    float rs = rsqrtf(q * (1.0f / N2) - mu * mu + 1e-5f);
#pragma unroll
    for (int nt = 0; nt < NT2; ++nt) {
      int n = nt * 16 + (lane & 15);
      acc2[nt][r] = (acc2[nt][r] - mu) * rs * g2[n] + bb2[n];
    }
  }
}

// ---------------------------------------------------------------------------
// Edge MLP layer: gather [x[row], x[col], ea] -> MLP -> ea += res; atomic scatter to agg
// ---------------------------------------------------------------------------
__global__ __launch_bounds__(128) void edge_mlp_kernel(
    const float* __restrict__ x, float* __restrict__ ea, float* __restrict__ agg,
    const int* __restrict__ ei,
    const _Float16* __restrict__ w1t, const float* __restrict__ b1,
    const float* __restrict__ g1, const float* __restrict__ bb1,
    const _Float16* __restrict__ w2t, const float* __restrict__ b2,
    const float* __restrict__ g2, const float* __restrict__ bb2) {
  extern __shared__ _Float16 lds[];  // 64*384 halves
  const int e0 = blockIdx.x * 64;
  // gather stage -> f16 LDS [64][384]
  for (int i = threadIdx.x; i < 64 * 384; i += 128) {
    int m = i / 384, k = i - m * 384;
    int e = e0 + m;
    float v;
    if (k < 128)        v = x[(size_t)ei[e] * Hh + k];
    else if (k < 256)   v = x[(size_t)ei[Ee + e] * Hh + (k - 128)];
    else                v = ea[(size_t)e * Hh + (k - 256)];
    lds[i] = (_Float16)v;
  }
  __syncthreads();
  v8f acc2[8];
  fused_mlp<384, 256, 128>(lds, w1t, b1, g1, bb1, w2t, b2, g2, bb2, acc2);
  // epilogue: residual update of ea (in place; this block owns these rows) + scatter-add
  const int lane = threadIdx.x & 31, wave = threadIdx.x >> 5;
  const int mb = wave * 16;
#pragma unroll
  for (int nt = 0; nt < 8; ++nt) {
    int n = nt * 16 + (lane & 15);
#pragma unroll
    for (int r = 0; r < 8; ++r) {
      int e = e0 + mb + r + ((lane >> 4) << 3);
      size_t idx = (size_t)e * Hh + n;
      float v = ea[idx] + acc2[nt][r];
      ea[idx] = v;
      int c = ei[Ee + e];
      atomicAdd(&agg[(size_t)c * Hh + n], v);
    }
  }
}

// ---------------------------------------------------------------------------
// Node MLP layer: gather [x, agg/cnt] -> MLP -> x += res (in place)
// ---------------------------------------------------------------------------
__global__ __launch_bounds__(128) void node_mlp_kernel(
    float* __restrict__ x, const float* __restrict__ agg, const float* __restrict__ cnt,
    const _Float16* __restrict__ w1t, const float* __restrict__ b1,
    const float* __restrict__ g1, const float* __restrict__ bb1,
    const _Float16* __restrict__ w2t, const float* __restrict__ b2,
    const float* __restrict__ g2, const float* __restrict__ bb2) {
  extern __shared__ _Float16 lds[];  // 64*256 halves
  const int i0 = blockIdx.x * 64;
  for (int i = threadIdx.x; i < 64 * 256; i += 128) {
    int m = i >> 8, k = i & 255;
    int nd = i0 + m;
    float v = 0.f;
    if (nd < Nn) {
      if (k < 128) v = x[(size_t)nd * Hh + k];
      else         v = agg[(size_t)nd * Hh + (k - 128)] / fmaxf(cnt[nd], 1.0f);
    }
    lds[i] = (_Float16)v;
  }
  __syncthreads();
  v8f acc2[8];
  fused_mlp<256, 256, 128>(lds, w1t, b1, g1, bb1, w2t, b2, g2, bb2, acc2);
  const int lane = threadIdx.x & 31, wave = threadIdx.x >> 5;
  const int mb = wave * 16;
#pragma unroll
  for (int nt = 0; nt < 8; ++nt) {
    int n = nt * 16 + (lane & 15);
#pragma unroll
    for (int r = 0; r < 8; ++r) {
      int nd = i0 + mb + r + ((lane >> 4) << 3);
      if (nd < Nn) {
        size_t idx = (size_t)nd * Hh + n;
        x[idx] = x[idx] + acc2[nt][r];
      }
    }
  }
}

// ---------------------------------------------------------------------------
// Small kernels
// ---------------------------------------------------------------------------
__global__ __launch_bounds__(256) void convert_transpose_kernel(
    const float* __restrict__ src, _Float16* __restrict__ dst, int K, int Nout) {
  int i = blockIdx.x * 256 + threadIdx.x;
  if (i < K * Nout) {
    int n = i % Nout, k = i / Nout;
    dst[(size_t)n * K + k] = (_Float16)src[(size_t)k * Nout + n];
  }
}

__global__ __launch_bounds__(256) void cnt_kernel(const int* __restrict__ ei,
                                                  float* __restrict__ cnt) {
  int e = blockIdx.x * 256 + threadIdx.x;
  if (e < Ee) atomicAdd(&cnt[ei[Ee + e]], 1.0f);
}

__global__ __launch_bounds__(128) void node_encoder_kernel(
    const float* __restrict__ x3, const float* __restrict__ w, const float* __restrict__ b,
    const float* __restrict__ g, const float* __restrict__ bb, float* __restrict__ h) {
  __shared__ float xr[3];
  __shared__ float ws[4], wq[4];
  const int i = blockIdx.x, n = threadIdx.x;
  if (n < 3) xr[n] = x3[(size_t)i * 3 + n];
  __syncthreads();
  float v = b[n];
#pragma unroll
  for (int k = 0; k < 3; ++k) v += xr[k] * w[k * Hh + n];
  float s = v, q = v * v;
#pragma unroll
  for (int m = 1; m <= 16; m <<= 1) { s += __shfl_xor(s, m, 32); q += __shfl_xor(q, m, 32); }
  const int wv = threadIdx.x >> 5, ln = threadIdx.x & 31;
  if (ln == 0) { ws[wv] = s; wq[wv] = q; }
  __syncthreads();
  s = ws[0] + ws[1] + ws[2] + ws[3];
  q = wq[0] + wq[1] + wq[2] + wq[3];
  float mu = s * (1.0f / 128.f), rs = rsqrtf(q * (1.0f / 128.f) - mu * mu + 1e-5f);
  h[(size_t)i * Hh + n] = gelu_f((v - mu) * rs * g[n] + bb[n]);
}

__global__ __launch_bounds__(128) void edge_encoder_kernel(
    const float* __restrict__ attr, const float* __restrict__ w, const float* __restrict__ b,
    float* __restrict__ ea) {
  __shared__ float ar[8];
  const int e = blockIdx.x, n = threadIdx.x;
  if (n < 8) ar[n] = attr[(size_t)e * 8 + n];
  __syncthreads();
  float v = b[n];
#pragma unroll
  for (int k = 0; k < 8; ++k) v += ar[k] * w[k * Hh + n];
  ea[(size_t)e * Hh + n] = v;
}

__global__ __launch_bounds__(128) void decoder_kernel(
    const float* __restrict__ h, const float* __restrict__ w1, const float* __restrict__ b1,
    const float* __restrict__ w2, const float* __restrict__ b2, float* __restrict__ out) {
  __shared__ float hr[128];
  __shared__ float hid[128];
  const int i = blockIdx.x, n = threadIdx.x;
  hr[n] = h[(size_t)i * Hh + n];
  __syncthreads();
  float v = b1[n];
  for (int k = 0; k < 128; ++k) v += hr[k] * w1[k * Hh + n];
  hid[n] = gelu_f(v);
  __syncthreads();
  if (n < 4) {
    float o = b2[n];
    for (int k = 0; k < 128; ++k) o += hid[k] * w2[k * 4 + n];
    out[(size_t)i * 4 + n] = o;
  }
}

// ---------------------------------------------------------------------------
// Host side
// ---------------------------------------------------------------------------
extern "C" void kernel_launch(void* const* d_in, const int* in_sizes, int n_in,
                              void* d_out, int out_size, void* d_ws, size_t ws_size,
                              hipStream_t stream) {
  (void)in_sizes; (void)n_in; (void)out_size; (void)ws_size;
  const float* x3     = (const float*)d_in[0];
  const int*   ei     = (const int*)d_in[1];
  const float* attr   = (const float*)d_in[2];
  const float* enc_w  = (const float*)d_in[3];
  const float* enc_b  = (const float*)d_in[4];
  const float* enc_g  = (const float*)d_in[5];
  const float* enc_bb = (const float*)d_in[6];
  const float* eenc_w = (const float*)d_in[7];
  const float* eenc_b = (const float*)d_in[8];
  const float* e_w1   = (const float*)d_in[9];
  const float* e_b1   = (const float*)d_in[10];
  const float* e_g1   = (const float*)d_in[11];
  const float* e_bb1  = (const float*)d_in[12];
  const float* e_w2   = (const float*)d_in[13];
  const float* e_b2   = (const float*)d_in[14];
  const float* e_g2   = (const float*)d_in[15];
  const float* e_bb2  = (const float*)d_in[16];
  const float* n_w1   = (const float*)d_in[17];
  const float* n_b1   = (const float*)d_in[18];
  const float* n_g1   = (const float*)d_in[19];
  const float* n_bb1  = (const float*)d_in[20];
  const float* n_w2   = (const float*)d_in[21];
  const float* n_b2   = (const float*)d_in[22];
  const float* n_g2   = (const float*)d_in[23];
  const float* n_bb2  = (const float*)d_in[24];
  const float* dec_w1 = (const float*)d_in[25];
  const float* dec_b1 = (const float*)d_in[26];
  const float* dec_w2 = (const float*)d_in[27];
  const float* dec_b2 = (const float*)d_in[28];
  float* out = (float*)d_out;

  // workspace carve-out (~363 MB total)
  char* ws = (char*)d_ws;
  float* h   = (float*)ws; ws += (size_t)Nn * Hh * sizeof(float);
  float* ea  = (float*)ws; ws += (size_t)Ee * Hh * sizeof(float);
  float* agg = (float*)ws; ws += (size_t)Nn * Hh * sizeof(float);
  float* cnt = (float*)ws; ws += (size_t)Nn * sizeof(float);
  _Float16* ew1t = (_Float16*)ws; ws += (size_t)Ll * 256 * 384 * sizeof(_Float16);
  _Float16* ew2t = (_Float16*)ws; ws += (size_t)Ll * 128 * 256 * sizeof(_Float16);
  _Float16* nw1t = (_Float16*)ws; ws += (size_t)Ll * 256 * 256 * sizeof(_Float16);
  _Float16* nw2t = (_Float16*)ws; ws += (size_t)Ll * 128 * 256 * sizeof(_Float16);

  // weight convert + transpose to [Nout][K] f16 (B-fragment friendly, L2-resident)
  for (int l = 0; l < Ll; ++l) {
    convert_transpose_kernel<<<(384 * 256 + 255) / 256, 256, 0, stream>>>(
        e_w1 + (size_t)l * 384 * 256, ew1t + (size_t)l * 256 * 384, 384, 256);
    convert_transpose_kernel<<<(256 * 128 + 255) / 256, 256, 0, stream>>>(
        e_w2 + (size_t)l * 256 * 128, ew2t + (size_t)l * 128 * 256, 256, 128);
    convert_transpose_kernel<<<(256 * 256 + 255) / 256, 256, 0, stream>>>(
        n_w1 + (size_t)l * 256 * 256, nw1t + (size_t)l * 256 * 256, 256, 256);
    convert_transpose_kernel<<<(256 * 128 + 255) / 256, 256, 0, stream>>>(
        n_w2 + (size_t)l * 256 * 128, nw2t + (size_t)l * 128 * 256, 256, 128);
  }

  // in-degree counts (constant across layers)
  hipMemsetAsync(cnt, 0, (size_t)Nn * sizeof(float), stream);
  cnt_kernel<<<(Ee + 255) / 256, 256, 0, stream>>>(ei, cnt);

  // encoders
  node_encoder_kernel<<<Nn, 128, 0, stream>>>(x3, enc_w, enc_b, enc_g, enc_bb, h);
  edge_encoder_kernel<<<Ee, 128, 0, stream>>>(attr, eenc_w, eenc_b, ea);

  // message-passing layers
  for (int l = 0; l < Ll; ++l) {
    hipMemsetAsync(agg, 0, (size_t)Nn * Hh * sizeof(float), stream);
    edge_mlp_kernel<<<Ee / 64, 128, 64 * 384 * sizeof(_Float16), stream>>>(
        h, ea, agg, ei,
        ew1t + (size_t)l * 256 * 384, e_b1 + (size_t)l * 256,
        e_g1 + (size_t)l * 256, e_bb1 + (size_t)l * 256,
        ew2t + (size_t)l * 128 * 256, e_b2 + (size_t)l * 128,
        e_g2 + (size_t)l * 128, e_bb2 + (size_t)l * 128);
    node_mlp_kernel<<<(Nn + 63) / 64, 128, 64 * 256 * sizeof(_Float16), stream>>>(
        h, agg, cnt,
        nw1t + (size_t)l * 256 * 256, n_b1 + (size_t)l * 256,
        n_g1 + (size_t)l * 256, n_bb1 + (size_t)l * 256,
        nw2t + (size_t)l * 128 * 256, n_b2 + (size_t)l * 128,
        n_g2 + (size_t)l * 128, n_bb2 + (size_t)l * 128);
  }

  // decoder
  decoder_kernel<<<Nn, 128, 0, stream>>>(h, dec_w1, dec_b1, dec_w2, dec_b2, out);
}
